// SelfAttentionBlock_1503238554223
// MI455X (gfx1250) — matrix operand order repaired and checked by reading, not compiled, not run on hardware
//
#include <hip/hip_runtime.h>
#include <cstddef>

// ---------- CDNA5 WMMA types ----------
typedef __attribute__((ext_vector_type(16))) __bf16 v16bf;
typedef __attribute__((ext_vector_type(8)))  __bf16 v8bf;
typedef __attribute__((ext_vector_type(8)))  float  v8f;

union AFrag { struct { v8bf lo, hi; } p; v16bf v; };

static __device__ inline v8f wmma_bf16(v16bf a, v16bf b, v8f c) {
  // D = A(16x32 bf16) * B(32x16 bf16) + C(16x16 f32)
  return __builtin_amdgcn_wmma_f32_16x16x32_bf16(false, a, false, b, (short)0, c,
                                                 false, false);
}

// A-fragment K index for half h (0..15): lanes 0-15 cover K 0-7,16-23; lanes 16-31 cover 8-15,24-31
static __device__ inline int a_k(int lane, int h) {
  return ((lane & 16) ? 8 : 0) + h + ((h & 8) ? 8 : 0);
}

// ---------- problem constants ----------
#define Cc   512
#define Bn   16
#define Npix 4096   // 64*64
#define Mkey 1024   // 32*32
#define Dq   64     // C/8
#define Vd   256    // C/2
#define Rall 384    // 64 + 64 + 256 fused projection rows

// ============ Kernel 0a: transpose+convert X -> Xt[b][n][c] bf16 ============
__global__ __launch_bounds__(256) void k_pack_xt(const float* __restrict__ x,
                                                 __bf16* __restrict__ Xt) {
  __shared__ float tile[32][33];
  int bx = blockIdx.x;                       // 128 ntiles * 16 ctiles * 16 batches
  int nt = bx & 127;
  int ct = (bx >> 7) & 15;
  int b  = bx >> 11;
  int tx = threadIdx.x & 31, ty = threadIdx.x >> 5;   // 32x8
  const float* xb = x + (size_t)b * Cc * Npix;
#pragma unroll
  for (int j = 0; j < 4; ++j)
    tile[ty + j * 8][tx] = xb[(size_t)(ct * 32 + ty + j * 8) * Npix + nt * 32 + tx];
  __syncthreads();
  __bf16* Xb = Xt + (size_t)b * Npix * Cc;
#pragma unroll
  for (int j = 0; j < 4; ++j)
    Xb[(size_t)(nt * 32 + ty + j * 8) * Cc + ct * 32 + tx] =
        (__bf16)tile[tx][ty + j * 8];
}

// ============ Kernel 0b: pack [w_theta;w_phi;w_g] into B-fragment-major bf16 ============
// Wpk[((kc*24 + rt)*32 + lane)*16 + h] = Wall[rt*16 + (lane&15)][kc*32 + a_k(lane,h)]
__global__ void k_pack_w(const float* __restrict__ wt, const float* __restrict__ wp,
                         const float* __restrict__ wg, __bf16* __restrict__ Wpk) {
  int tid  = blockIdx.x * blockDim.x + threadIdx.x;   // 16*24*32*16 = 196608
  int h    = tid & 15;
  int lane = (tid >> 4) & 31;
  int rt   = (tid >> 9) % 24;
  int kc   = (tid >> 9) / 24;
  int row  = rt * 16 + (lane & 15);
  int k    = kc * 32 + a_k(lane, h);
  const float* wr = (row < 64)  ? (wt + (size_t)row * Cc)
                  : (row < 128) ? (wp + (size_t)(row - 64) * Cc)
                                : (wg + (size_t)(row - 128) * Cc);
  Wpk[tid] = (__bf16)wr[k];
}

// ============ Kernel 0c: pack w_o into A-fragment-major bf16 ============
// Wopk[((kc*32 + ct)*32 + lane)*16 + h] = wo[ct*16 + (lane&15)][kc*32 + ((lane&16)?16:0) + h]
__global__ void k_pack_wo(const float* __restrict__ wo, __bf16* __restrict__ Wopk) {
  int tid  = blockIdx.x * blockDim.x + threadIdx.x;   // 8*32*32*16 = 131072
  int h    = tid & 15;
  int lane = (tid >> 4) & 31;
  int ct   = (tid >> 9) & 31;
  int kc   = tid >> 14;
  Wopk[tid] = (__bf16)wo[(size_t)(ct * 16 + (lane & 15)) * Vd + kc * 32 + ((lane & 16) ? 16 : 0) + h];
}

// ============ Kernel 1: F2[b][n][r] = Xt[b] * Wall^T (bf16 WMMA) ============
__global__ __launch_bounds__(256) void k_gemm_xw(
    const __bf16* __restrict__ Xt, const __bf16* __restrict__ Wpk,
    float* __restrict__ F2) {
  int lane = threadIdx.x & 31;
  int wid  = blockIdx.x * 8 + (threadIdx.x >> 5);   // 98304 waves = 16 * 24 * 256
  int nt   = wid & 255;
  int tmp  = wid >> 8;
  int rt   = tmp % 24;
  int b    = tmp / 24;

  int akb = (lane & 16) ? 8 : 0;
  const __bf16* Xrow = Xt + ((size_t)b * Npix + nt * 16 + (lane & 15)) * Cc;

  v8f acc = {};
#pragma unroll 4
  for (int kc = 0; kc < 16; ++kc) {
    int k0 = kc * 32;
    AFrag a;
    a.p.lo = *(const v8bf*)(Xrow + k0 + akb);
    a.p.hi = *(const v8bf*)(Xrow + k0 + 16 + akb);
    v16bf bf = *(const v16bf*)(Wpk + (((size_t)kc * 24 + rt) * 32 + lane) * 16);
    acc = wmma_bf16(a.v, bf, acc);
  }

  int rowoff = (lane & 16) ? 8 : 0;
  float* Fb = F2 + ((size_t)b * Npix + nt * 16) * Rall + rt * 16 + (lane & 15);
#pragma unroll
  for (int r = 0; r < 8; ++r)
    Fb[(size_t)(rowoff + r) * Rall] = acc[r];
}

// ============ Kernel 2: pack Q / pooled K / pooled Vt (bf16) ============
__global__ void k_pack_q(const float* __restrict__ F2, __bf16* __restrict__ Qb) {
  int tid = blockIdx.x * blockDim.x + threadIdx.x;   // 16*4096*64
  int d = tid & 63, n = (tid >> 6) & 4095, b = tid >> 18;
  Qb[tid] = (__bf16)F2[((size_t)b * Npix + n) * Rall + d];
}

__global__ void k_pack_k(const float* __restrict__ F2, __bf16* __restrict__ Kb) {
  int tid = blockIdx.x * blockDim.x + threadIdx.x;   // 16*1024*64
  int d = tid & 63, m = (tid >> 6) & 1023, b = tid >> 16;
  int py = m >> 5, px = m & 31;
  int n00 = (py * 2) * 64 + px * 2;
  const float* Fb = F2 + (size_t)b * Npix * Rall;
  float v = fmaxf(fmaxf(Fb[(size_t)n00 * Rall + 64 + d],
                        Fb[(size_t)(n00 + 1) * Rall + 64 + d]),
                  fmaxf(Fb[(size_t)(n00 + 64) * Rall + 64 + d],
                        Fb[(size_t)(n00 + 65) * Rall + 64 + d]));
  Kb[tid] = (__bf16)v;
}

// Vt[b][v][m] transposed so attention PV B-fragments load contiguous along m.
__global__ void k_pack_v(const float* __restrict__ F2, __bf16* __restrict__ Vt) {
  int tid = blockIdx.x * blockDim.x + threadIdx.x;   // 16*256*1024
  int m = tid & 1023, v = (tid >> 10) & 255, b = tid >> 18;
  int py = m >> 5, px = m & 31;
  int n00 = (py * 2) * 64 + px * 2;
  const float* Fb = F2 + (size_t)b * Npix * Rall;
  float val = fmaxf(fmaxf(Fb[(size_t)n00 * Rall + 128 + v],
                          Fb[(size_t)(n00 + 1) * Rall + 128 + v]),
                    fmaxf(Fb[(size_t)(n00 + 64) * Rall + 128 + v],
                          Fb[(size_t)(n00 + 65) * Rall + 128 + v]));
  Vt[tid] = (__bf16)val;
}

// ============ Kernel 3: flash attention ============
// Q [B][N][64], K [B][M][64], Vt [B][256][M], O [B][N][256]; all bf16.
__global__ __launch_bounds__(256) void k_attn(
    const __bf16* __restrict__ Qb, const __bf16* __restrict__ Kb,
    const __bf16* __restrict__ Vt, __bf16* __restrict__ Ob) {
  __shared__ float sP[8][16 * 32];                  // per-wave C->A relayout slab
  int w    = threadIdx.x >> 5;
  int lane = threadIdx.x & 31;
  int wid  = blockIdx.x * 8 + w;                    // 16384 waves = 16 * 256 * 4
  int vc   = wid & 3;
  int qt   = (wid >> 2) & 255;
  int b    = wid >> 10;

  int akb = (lane & 16) ? 8 : 0;
  int bkb = (lane & 16) ? 16 : 0;
  int col = lane & 15;

  // Q fragments (loop-invariant): rows qt*16+M, d 0..63 as two K=32 chunks
  const __bf16* Qr = Qb + ((size_t)(b * Npix) + qt * 16 + col) * Dq;
  AFrag aq0, aq1;
  aq0.p.lo = *(const v8bf*)(Qr + bkb);
  aq0.p.hi = *(const v8bf*)(Qr + 8 + bkb);
  aq1.p.lo = *(const v8bf*)(Qr + 32 + bkb);
  aq1.p.hi = *(const v8bf*)(Qr + 40 + bkb);

  float rmax[8], rsum[8];
  v8f oacc[4];
  v8f zero = {};
#pragma unroll
  for (int r = 0; r < 8; ++r) { rmax[r] = -1e30f; rsum[r] = 0.f; }
#pragma unroll
  for (int t = 0; t < 4; ++t) oacc[t] = zero;

  float* sw = sP[w];

  for (int ms = 0; ms < Mkey; ms += 32) {
    // ---- scores: two 16x16 tiles (keys ms..ms+15, ms+16..ms+31) ----
    const __bf16* K0 = Kb + ((size_t)(b * Mkey) + ms + col) * Dq;
    const __bf16* K1 = K0 + 16 * Dq;
    v16bf b0a = *(const v16bf*)(K0 + bkb);
    v16bf b0b = *(const v16bf*)(K0 + 32 + bkb);
    v16bf b1a = *(const v16bf*)(K1 + bkb);
    v16bf b1b = *(const v16bf*)(K1 + 32 + bkb);
    v8f s0 = wmma_bf16(aq0.v, b0a, zero);
    s0     = wmma_bf16(aq1.v, b0b, s0);
    v8f s1 = wmma_bf16(aq0.v, b1a, zero);
    s1     = wmma_bf16(aq1.v, b1b, s1);

    // ---- online softmax row stats (row = lane-group dependent) ----
    float p0[8], p1[8], corr[8];
#pragma unroll
    for (int r = 0; r < 8; ++r) {
      float lm = fmaxf(s0[r], s1[r]);
#pragma unroll
      for (int msk = 1; msk < 16; msk <<= 1) lm = fmaxf(lm, __shfl_xor(lm, msk, 32));
      float nm = fmaxf(rmax[r], lm);
      corr[r]  = __expf(rmax[r] - nm);
      float e0 = __expf(s0[r] - nm);
      float e1 = __expf(s1[r] - nm);
      float ls = e0 + e1;
#pragma unroll
      for (int msk = 1; msk < 16; msk <<= 1) ls += __shfl_xor(ls, msk, 32);
      rsum[r] = rsum[r] * corr[r] + ls;
      rmax[r] = nm;
      p0[r] = e0;
      p1[r] = e1;
    }
#pragma unroll
    for (int t = 0; t < 4; ++t)
#pragma unroll
      for (int r = 0; r < 8; ++r) oacc[t][r] *= corr[r];

    // ---- redistribute P: C layout -> A layout through LDS ----
    int rowoff = (lane & 16) ? 8 : 0;
#pragma unroll
    for (int r = 0; r < 8; ++r) {
      sw[(rowoff + r) * 32 + col]      = p0[r];
      sw[(rowoff + r) * 32 + 16 + col] = p1[r];
    }
    __syncthreads();
    v16bf ps;
#pragma unroll
    for (int h = 0; h < 16; ++h)
      ps[h] = (__bf16)sw[col * 32 + bkb + h];
    __syncthreads();

    // ---- O += P * V (4 tiles of 16 v-columns); Vt gives contiguous K loads ----
#pragma unroll
    for (int t = 0; t < 4; ++t) {
      int vcol = vc * 64 + t * 16 + col;
      v16bf bvv = *(const v16bf*)(Vt + ((size_t)(b * Vd) + vcol) * Mkey + ms + bkb);
      oacc[t] = wmma_bf16(ps, bvv, oacc[t]);
    }
  }

  // ---- normalize and store O bf16 [B][N][256] ----
  int rowoff = (lane & 16) ? 8 : 0;
#pragma unroll
  for (int t = 0; t < 4; ++t) {
    int vcol = vc * 64 + t * 16 + col;
#pragma unroll
    for (int r = 0; r < 8; ++r) {
      int n = qt * 16 + rowoff + r;
      Ob[((size_t)(b * Npix) + n) * Vd + vcol] = (__bf16)(oacc[t][r] / rsum[r]);
    }
  }
}

// ============ Kernel 4: out = gamma * (w_o * O^T) + x ============
__global__ __launch_bounds__(256) void k_proj(
    const __bf16* __restrict__ Wopk, const __bf16* __restrict__ Ob,
    const float* __restrict__ x, const float* __restrict__ gamma,
    float* __restrict__ out) {
  int lane = threadIdx.x & 31;
  int wid  = blockIdx.x * 8 + (threadIdx.x >> 5);   // 131072 = 16*32*256
  int nt   = wid & 255;
  int ct   = (wid >> 8) & 31;
  int b    = wid >> 13;

  int bkb = (lane & 16) ? 16 : 0;
  int col = lane & 15;

  const __bf16* Orow = Ob + ((size_t)(b * Npix) + nt * 16 + col) * Vd;

  v8f acc = {};
#pragma unroll
  for (int kc = 0; kc < 8; ++kc) {
    v16bf av = *(const v16bf*)(Wopk + (((size_t)kc * 32 + ct) * 32 + lane) * 16);
    v16bf bv = *(const v16bf*)(Orow + kc * 32 + bkb);   // 32B contiguous bf16
    acc = wmma_bf16(av, bv, acc);
  }

  float gm = *gamma;
  int rowoff = (lane & 16) ? 8 : 0;
#pragma unroll
  for (int r = 0; r < 8; ++r) {
    size_t idx = ((size_t)b * Cc + ct * 16 + rowoff + r) * Npix + nt * 16 + col;
    out[idx] = gm * acc[r] + x[idx];
  }
}

// ================= launch =================
extern "C" void kernel_launch(void* const* d_in, const int* in_sizes, int n_in,
                              void* d_out, int out_size, void* d_ws, size_t ws_size,
                              hipStream_t stream) {
  (void)in_sizes; (void)n_in; (void)out_size; (void)ws_size;
  const float* x     = (const float*)d_in[0];
  const float* wt    = (const float*)d_in[1];
  const float* wp    = (const float*)d_in[2];
  const float* wg    = (const float*)d_in[3];
  const float* wo    = (const float*)d_in[4];
  const float* gamma = (const float*)d_in[5];
  float* out = (float*)d_out;

  char* ws = (char*)d_ws;
  __bf16* Xt   = (__bf16*)(ws);                      //  67,108,864 B
  float*  F2   = (float*)(ws + 67108864);            // 100,663,296 B
  __bf16* Qb   = (__bf16*)(ws + 167772160);          //   8,388,608 B
  __bf16* Kb   = (__bf16*)(ws + 176160768);          //   2,097,152 B
  __bf16* Vt   = (__bf16*)(ws + 178257920);          //   8,388,608 B
  __bf16* Ob   = (__bf16*)(ws + 186646528);          //  33,554,432 B
  __bf16* Wpk  = (__bf16*)(ws + 220200960);          //     393,216 B
  __bf16* Wopk = (__bf16*)(ws + 220594176);          //     262,144 B

  k_pack_xt<<<32768, 256, 0, stream>>>(x, Xt);
  k_pack_w<<<768, 256, 0, stream>>>(wt, wp, wg, Wpk);
  k_pack_wo<<<512, 256, 0, stream>>>(wo, Wopk);
  k_gemm_xw<<<12288, 256, 0, stream>>>(Xt, Wpk, F2);
  k_pack_q<<<16384, 256, 0, stream>>>(F2, Qb);
  k_pack_k<<<4096, 256, 0, stream>>>(F2, Kb);
  k_pack_v<<<16384, 256, 0, stream>>>(F2, Vt);
  k_attn<<<2048, 256, 0, stream>>>(Qb, Kb, Vt, Ob);
  k_proj<<<16384, 256, 0, stream>>>(Wopk, Ob, x, gamma, out);
}
